// Block_2783138808310
// MI455X (gfx1250) — matrix operand order repaired and checked by reading, not compiled, not run on hardware
//
#include <hip/hip_runtime.h>
#include <hip/hip_bf16.h>
#include <math.h>

// ---------------------------------------------------------------------------
// CDNA5 (gfx1250) wave32 WMMA implementation of the deformable-DETR block.
// All GEMMs + attention matmuls use v_wmma_f32_16x16x32_bf16 (fp32 accum).
// GEMM waves compute 32x64 tiles (8 WMMA / 32-K step). Attention uses an
// LDS-staged online softmax with zero cross-lane shuffles (no ds_bpermute).
// ---------------------------------------------------------------------------

typedef __bf16 bf16;
typedef __attribute__((ext_vector_type(16))) __bf16 v16bf;
typedef __attribute__((ext_vector_type(8)))  __bf16 v8bf;
typedef __attribute__((ext_vector_type(8)))  float  v8f;
typedef __attribute__((ext_vector_type(4)))  float  v4f;

// ---- WMMA fragment helpers (layouts per CDNA5 ISA 7.12.2, wave32) ----------

// A-matrix 16x32 bf16: lane L -> row M=L&15; lanes 0-15 hold K {0..7,16..23},
// lanes 16-31 hold K {8..15,24..31}.
__device__ __forceinline__ v16bf load_a_frag(const bf16* A, int lda, int row0,
                                             int k0, int lane) {
  int m  = row0 + (lane & 15);
  int kb = k0 + ((lane >> 4) << 3);
  const bf16* p = A + (size_t)m * lda + kb;
  v8bf lo = *reinterpret_cast<const v8bf*>(p);
  v8bf hi = *reinterpret_cast<const v8bf*>(p + 16);
  v16bf r;
#pragma unroll
  for (int i = 0; i < 8; ++i) { r[i] = lo[i]; r[i + 8] = hi[i]; }
  return r;
}

// B-matrix 32x16 bf16: lane L -> col N=L&15; lanes 0-15 hold K {0..7,16..23},
// lanes 16-31 hold K {8..15,24..31} (same K order as A). Bm is (N x K) row-major.
__device__ __forceinline__ v16bf load_b_frag(const bf16* Bm, int ldb, int n0,
                                             int k0, int lane) {
  int n  = n0 + (lane & 15);
  int kb = k0 + ((lane >> 4) << 3);
  const bf16* p = Bm + (size_t)n * ldb + kb;
  v8bf lo = *reinterpret_cast<const v8bf*>(p);
  v8bf hi = *reinterpret_cast<const v8bf*>(p + 16);
  v16bf r;
#pragma unroll
  for (int i = 0; i < 8; ++i) { r[i] = lo[i]; r[i + 8] = hi[i]; }
  return r;
}

__device__ __forceinline__ v8f wmma_bf16(v16bf a, v16bf b, v8f c) {
  return __builtin_amdgcn_wmma_f32_16x16x32_bf16(false, a, false, b, (short)0,
                                                 c, false, false);
}

// ---- fp32 -> bf16 conversion ----------------------------------------------

__global__ __launch_bounds__(256) void f32_to_bf16_kernel(
    const float* __restrict__ in, bf16* __restrict__ out, int n) {
  int i = blockIdx.x * blockDim.x + threadIdx.x;
  if (i < n) out[i] = (bf16)in[i];
}

// ---- LayerNorm (C=256), one wave per row, bf16 output ---------------------

__global__ __launch_bounds__(256) void ln_bf16_kernel(
    const float* __restrict__ x, const float* __restrict__ w,
    const float* __restrict__ b, bf16* __restrict__ out, int rows) {
  int wid = blockIdx.x * 8 + (threadIdx.x >> 5);
  if (wid >= rows) return;
  int lane = threadIdx.x & 31;
  const float* xr = x + (size_t)wid * 256;
  float v[8];
  float s = 0.f;
#pragma unroll
  for (int i = 0; i < 8; ++i) { v[i] = xr[i * 32 + lane]; s += v[i]; }
#pragma unroll
  for (int m = 16; m; m >>= 1) s += __shfl_xor(s, m, 32);
  float mean = s * (1.f / 256.f);
  float var = 0.f;
#pragma unroll
  for (int i = 0; i < 8; ++i) { float d = v[i] - mean; var += d * d; }
#pragma unroll
  for (int m = 16; m; m >>= 1) var += __shfl_xor(var, m, 32);
  float rstd = rsqrtf(var * (1.f / 256.f) + 1e-5f);
  bf16* orow = out + (size_t)wid * 256;
#pragma unroll
  for (int i = 0; i < 8; ++i) {
    int c = i * 32 + lane;
    orow[c] = (bf16)((v[i] - mean) * rstd * w[c] + b[c]);
  }
}

// ---- Generic WMMA GEMM:  out = op(A(bf16,MxK) * W(bf16,NxK)^T + bias) [+resid]
// One wave computes a (16*TM)x(16*TN) tile. op: 0 = none, 1 = exact GELU.

template <int TM, int TN>
__global__ __launch_bounds__(256) void gemm_bf16_kernel(
    const bf16* __restrict__ A, const bf16* __restrict__ W,
    const float* __restrict__ bias, const float* __restrict__ resid,
    float* __restrict__ outF, bf16* __restrict__ outB, int M, int Nv, int K,
    int op) {
  int lane = threadIdx.x & 31;
  int wid = blockIdx.x * 8 + (threadIdx.x >> 5);
  int nblk = Nv / (16 * TN);
  int total = (M / (16 * TM)) * nblk;
  if (wid >= total) return;
  int mt = wid / nblk;
  int nt = wid - mt * nblk;

  v8f acc[TM][TN];
#pragma unroll
  for (int i = 0; i < TM; ++i)
#pragma unroll
    for (int j = 0; j < TN; ++j) acc[i][j] = {};

  const bf16* apre = A + (size_t)(mt * 16 * TM + (lane & 15)) * K;
  for (int k0 = 0; k0 < K; k0 += 32) {
    __builtin_prefetch(apre + k0 + 128, 0, 1);  // -> global_prefetch_b8
    v16bf af[TM];
#pragma unroll
    for (int i = 0; i < TM; ++i)
      af[i] = load_a_frag(A, K, (mt * TM + i) << 4, k0, lane);
    v16bf bf[TN];
#pragma unroll
    for (int j = 0; j < TN; ++j)
      bf[j] = load_b_frag(W, K, (nt * TN + j) << 4, k0, lane);
#pragma unroll
    for (int i = 0; i < TM; ++i)
#pragma unroll
      for (int j = 0; j < TN; ++j) acc[i][j] = wmma_bf16(af[i], bf[j], acc[i][j]);
  }

  int rofs = (lane >> 4) << 3;  // C/D layout: M = v + 8*(L>>4)
#pragma unroll
  for (int j = 0; j < TN; ++j) {
    int n = ((nt * TN + j) << 4) + (lane & 15);
    float bv = bias ? bias[n] : 0.f;
#pragma unroll
    for (int i = 0; i < TM; ++i) {
      int rbase = ((mt * TM + i) << 4) + rofs;
#pragma unroll
      for (int v = 0; v < 8; ++v) {
        size_t idx = (size_t)(rbase + v) * Nv + n;
        float val = acc[i][j][v] + bv;
        if (op == 1) val = 0.5f * val * (1.f + erff(val * 0.70710678118654752f));
        if (resid) val += resid[idx];
        if (outF) outF[idx] = val;
        if (outB) outB[idx] = (bf16)val;
      }
    }
  }
}

// ---- V transpose: vt[b][h][d][n] = qkv[b][n][2][h][d]  (bf16) --------------

__global__ __launch_bounds__(256) void vt_kernel(const bf16* __restrict__ qkv,
                                                 bf16* __restrict__ vt) {
  int i = blockIdx.x * blockDim.x + threadIdx.x;  // B*NH*DH*N = 4194304 exact
  int n = i & 1023;
  int d = (i >> 10) & 31;
  int h = (i >> 15) & 7;
  int b = i >> 18;
  vt[i] = qkv[(size_t)(b * 1024 + n) * 768 + 512 + h * 32 + d];
}

// ---- Flash attention (wave32 WMMA, LDS-staged online softmax) --------------
// One wave per (b, h, 32-row q block). Row statistics are computed by "row
// owner" lanes on LDS-staged score rows: no ds_bpermute anywhere.

__global__ __launch_bounds__(256) void attn_kernel(
    const bf16* __restrict__ qkv, const bf16* __restrict__ vt,
    bf16* __restrict__ out) {
  constexpr int SP = 36;  // fp32 pitch of an S row (padded vs 32)
  constexpr int PP = 40;  // bf16 pitch of a P row (padded vs 32)
  __shared__ float smemS[8 * 2 * 16 * SP];
  __shared__ bf16  smemP[8 * 2 * 16 * PP];
  __shared__ float smemT[8 * 32];

  int lane = threadIdx.x & 31;
  int wslot = threadIdx.x >> 5;
  int wid = blockIdx.x * 8 + wslot;  // total = 16*8*32 = 4096 exact
  int qb = wid & 31;                 // 32-row q block
  int h = (wid >> 5) & 7;
  int b = wid >> 8;

  float* S = smemS + wslot * (2 * 16 * SP);
  bf16*  P = smemP + wslot * (2 * 16 * PP);
  float* T = smemT + wslot * 32;

  const bf16* Qb = qkv + (size_t)b * 1024 * 768 + h * 32;
  const bf16* Kb = qkv + (size_t)b * 1024 * 768 + 256 + h * 32;
  const bf16* Vt = vt + (size_t)(b * 8 + h) * 32 * 1024;

  v16bf qa[2];
#pragma unroll
  for (int t = 0; t < 2; ++t)
    qa[t] = load_a_frag(Qb, 768, (qb << 5) + (t << 4), 0, lane);
  const float scale = 0.17677669529663687f;  // 1/sqrt(32)

  // Row-owner state: this lane owns row (lane&15) of tile (lane>>4).
  int st = lane >> 4, sr = lane & 15;
  float* Srow = S + (st * 16 + sr) * SP;
  bf16*  Prow = P + (st * 16 + sr) * PP;
  float mrow = -INFINITY, lrow = 0.f;

  v8f o0[2], o1[2];
  o0[0] = {}; o0[1] = {}; o1[0] = {}; o1[1] = {};
  int col = lane & 15;
  int rofs = (lane >> 4) << 3;

  for (int kb = 0; kb < 1024; kb += 32) {
    // S = Q K^T for 32 keys: reduction dim = DH = 32, K-frags reused 2x
    v16bf kf0 = load_b_frag(Kb, 768, kb, 0, lane);
    v16bf kf1 = load_b_frag(Kb, 768, kb + 16, 0, lane);
    v8f s0[2], s1[2];
#pragma unroll
    for (int t = 0; t < 2; ++t) {
      v8f z0 = {}, z1 = {};
      s0[t] = wmma_bf16(qa[t], kf0, z0);
      s1[t] = wmma_bf16(qa[t], kf1, z1);
    }
    // Scatter scaled scores (C/D layout) into row-major LDS rows.
#pragma unroll
    for (int t = 0; t < 2; ++t) {
      float* tS = S + t * 16 * SP;
#pragma unroll
      for (int v = 0; v < 8; ++v) {
        int row = v + rofs;
        tS[row * SP + col] = s0[t][v] * scale;
        tS[row * SP + 16 + col] = s1[t][v] * scale;
      }
    }
    asm volatile("s_wait_dscnt 0" ::: "memory");

    // Row owner: serial max / exp / sum over its 32 scores, emit bf16 P row.
    v4f sv[8];
#pragma unroll
    for (int i = 0; i < 8; ++i)
      sv[i] = *reinterpret_cast<const v4f*>(Srow + i * 4);
    float rmax = -INFINITY;
#pragma unroll
    for (int i = 0; i < 8; ++i)
#pragma unroll
      for (int j = 0; j < 4; ++j) rmax = fmaxf(rmax, sv[i][j]);
    float mnew = fmaxf(mrow, rmax);
    float corr = __expf(mrow - mnew);
    mrow = mnew;
    float rs = 0.f;
    v8bf pv[4];
#pragma unroll
    for (int i = 0; i < 4; ++i)
#pragma unroll
      for (int j = 0; j < 8; ++j) {
        int n = i * 8 + j;
        float e = __expf(sv[n >> 2][n & 3] - mnew);
        rs += e;
        pv[i][j] = (bf16)e;
      }
    lrow = lrow * corr + rs;
#pragma unroll
    for (int i = 0; i < 4; ++i)
      *reinterpret_cast<v8bf*>(Prow + i * 8) = pv[i];
    T[st * 16 + sr] = corr;
    asm volatile("s_wait_dscnt 0" ::: "memory");

    // Rescale O by per-row corr (broadcast via LDS, aligned b128 loads).
#pragma unroll
    for (int t = 0; t < 2; ++t) {
      v4f c0 = *reinterpret_cast<const v4f*>(T + t * 16 + rofs);
      v4f c1 = *reinterpret_cast<const v4f*>(T + t * 16 + rofs + 4);
#pragma unroll
      for (int v = 0; v < 4; ++v) {
        o0[t][v] *= c0[v];     o1[t][v] *= c0[v];
        o0[t][v + 4] *= c1[v]; o1[t][v + 4] *= c1[v];
      }
    }

    // O += P V : P from LDS as A-frag, V B-frags reused across both q tiles.
    v16bf vb0 = load_b_frag(Vt, 1024, 0, kb, lane);
    v16bf vb1 = load_b_frag(Vt, 1024, 16, kb, lane);
#pragma unroll
    for (int t = 0; t < 2; ++t) {
      const bf16* pp = P + (t * 16 + (lane & 15)) * PP + ((lane >> 4) << 3);
      v8bf lo = *reinterpret_cast<const v8bf*>(pp);
      v8bf hi = *reinterpret_cast<const v8bf*>(pp + 16);
      v16bf pa;
#pragma unroll
      for (int i = 0; i < 8; ++i) { pa[i] = lo[i]; pa[i + 8] = hi[i]; }
      o0[t] = wmma_bf16(pa, vb0, o0[t]);
      o1[t] = wmma_bf16(pa, vb1, o1[t]);
    }
  }

  // Final 1/l normalization, broadcast via LDS like corr.
  T[st * 16 + sr] = 1.f / lrow;
  asm volatile("s_wait_dscnt 0" ::: "memory");
#pragma unroll
  for (int t = 0; t < 2; ++t) {
    v4f c0 = *reinterpret_cast<const v4f*>(T + t * 16 + rofs);
    v4f c1 = *reinterpret_cast<const v4f*>(T + t * 16 + rofs + 4);
    int qrow0 = (qb << 5) + (t << 4) + rofs;
#pragma unroll
    for (int v = 0; v < 8; ++v) {
      float inv = (v < 4) ? c0[v] : c1[v - 4];
      size_t base = (size_t)(b * 1024 + qrow0 + v) * 256 + h * 32;
      out[base + col] = (bf16)(o0[t][v] * inv);
      out[base + 16 + col] = (bf16)(o1[t][v] * inv);
    }
  }
}

// ---- Deformable sampling: softmax(P=4) + bilinear gather -------------------
// One wave per (token m, head h); lane = channel d.

__device__ __forceinline__ float dsamp(const float* vh, int xi, int yi) {
  bool valid = (xi >= 0) & (xi < 32) & (yi >= 0) & (yi < 32);
  int xc = min(max(xi, 0), 31);
  int yc = min(max(yi, 0), 31);
  float val = vh[(size_t)(yc * 32 + xc) * 256];
  return valid ? val : 0.f;
}

__global__ __launch_bounds__(256) void deform_kernel(
    const float* __restrict__ ref, const float* __restrict__ vdef,
    const float* __restrict__ off, const float* __restrict__ awl,
    bf16* __restrict__ out) {
  int lane = threadIdx.x & 31;
  int wid = blockIdx.x * 8 + (threadIdx.x >> 5);  // B*N*NHD = 131072 exact
  int h = wid & 7;
  int m = wid >> 3;
  int b = m >> 10;

  float rx = ref[m * 2 + 0];
  float ry = ref[m * 2 + 1];

  const float* aw4 = awl + (size_t)m * 32 + h * 4;
  float a0 = aw4[0], a1 = aw4[1], a2 = aw4[2], a3 = aw4[3];
  float mx = fmaxf(fmaxf(a0, a1), fmaxf(a2, a3));
  float e0 = __expf(a0 - mx), e1 = __expf(a1 - mx);
  float e2 = __expf(a2 - mx), e3 = __expf(a3 - mx);
  float inv = 1.f / (e0 + e1 + e2 + e3);
  float wp[4] = {e0 * inv, e1 * inv, e2 * inv, e3 * inv};

  const float* offp = off + (size_t)m * 64 + h * 8;
  const float* vh = vdef + ((size_t)b * 1024 * 8 + h) * 32 + lane;

  float acc = 0.f;
#pragma unroll
  for (int p = 0; p < 4; ++p) {
    float gx = (rx + offp[p * 2 + 0] * (1.f / 32.f)) * 32.f - 0.5f;
    float gy = (ry + offp[p * 2 + 1] * (1.f / 32.f)) * 32.f - 0.5f;
    float x0f = floorf(gx), y0f = floorf(gy);
    float lx = gx - x0f, ly = gy - y0f;
    int x0 = (int)x0f, y0 = (int)y0f;
    float s = dsamp(vh, x0, y0) * (1.f - lx) * (1.f - ly) +
              dsamp(vh, x0 + 1, y0) * lx * (1.f - ly) +
              dsamp(vh, x0, y0 + 1) * (1.f - lx) * ly +
              dsamp(vh, x0 + 1, y0 + 1) * lx * ly;
    acc += wp[p] * s;
  }
  out[(size_t)m * 256 + h * 32 + lane] = (bf16)acc;
}

// ---------------------------------------------------------------------------

extern "C" void kernel_launch(void* const* d_in, const int* in_sizes, int n_in,
                              void* d_out, int out_size, void* d_ws,
                              size_t ws_size, hipStream_t stream) {
  const float* x      = (const float*)d_in[0];
  const float* ref    = (const float*)d_in[1];
  const float* value  = (const float*)d_in[2];
  const float* ln1_w  = (const float*)d_in[3];
  const float* ln1_b  = (const float*)d_in[4];
  const float* ln2_w  = (const float*)d_in[5];
  const float* ln2_b  = (const float*)d_in[6];
  const float* ln3_w  = (const float*)d_in[7];
  const float* ln3_b  = (const float*)d_in[8];
  const float* qkv_w  = (const float*)d_in[9];
  const float* proj_w = (const float*)d_in[10];
  const float* proj_b = (const float*)d_in[11];
  const float* off_w  = (const float*)d_in[12];
  const float* off_b  = (const float*)d_in[13];
  const float* aw_w   = (const float*)d_in[14];
  const float* aw_b   = (const float*)d_in[15];
  const float* vproj_w = (const float*)d_in[16];
  const float* vproj_b = (const float*)d_in[17];
  const float* oproj_w = (const float*)d_in[18];
  const float* oproj_b = (const float*)d_in[19];
  const float* fc1_w  = (const float*)d_in[20];
  const float* fc1_b  = (const float*)d_in[21];
  const float* fc2_w  = (const float*)d_in[22];
  const float* fc2_b  = (const float*)d_in[23];

  const int M = 16 * 1024;  // B*N
  const int C = 256, HID = 1024;

  char* ws = (char*)d_ws;
  size_t cur = 0;
  auto alloc = [&](size_t bytes) -> void* {
    void* p = ws + cur;
    cur = (cur + bytes + 255) & ~(size_t)255;
    return p;
  };

  bf16* wqkv  = (bf16*)alloc((size_t)768 * 256 * 2);
  bf16* wproj = (bf16*)alloc((size_t)256 * 256 * 2);
  bf16* woff  = (bf16*)alloc((size_t)64 * 256 * 2);
  bf16* waw   = (bf16*)alloc((size_t)32 * 256 * 2);
  bf16* wvprj = (bf16*)alloc((size_t)256 * 256 * 2);
  bf16* woprj = (bf16*)alloc((size_t)256 * 256 * 2);
  bf16* wfc1  = (bf16*)alloc((size_t)1024 * 256 * 2);
  bf16* wfc2  = (bf16*)alloc((size_t)256 * 1024 * 2);
  bf16* val_b = (bf16*)alloc((size_t)M * C * 2);
  bf16* ln_b  = (bf16*)alloc((size_t)M * C * 2);
  bf16* qkv_b = (bf16*)alloc((size_t)M * 768 * 2);
  bf16* vt_b  = (bf16*)alloc((size_t)M * C * 2);
  bf16* ao_b  = (bf16*)alloc((size_t)M * C * 2);  // attn out, reused as deform out
  bf16* h_b   = (bf16*)alloc((size_t)M * HID * 2);
  float* x1   = (float*)alloc((size_t)M * C * 4);
  float* x2   = (float*)alloc((size_t)M * C * 4);
  float* vdef = (float*)alloc((size_t)M * C * 4);
  float* offf = (float*)alloc((size_t)M * 64 * 4);
  float* awlf = (float*)alloc((size_t)M * 32 * 4);
  (void)ws_size; (void)in_sizes; (void)n_in; (void)out_size;

  auto cvt = [&](const float* s, bf16* d, int n) {
    f32_to_bf16_kernel<<<(n + 255) / 256, 256, 0, stream>>>(s, d, n);
  };
  auto gemm = [&](const bf16* A, const bf16* W, const float* bias,
                  const float* resid, float* oF, bf16* oB, int M_, int N_,
                  int K_, int op) {
    if (N_ % 64 == 0) {
      int waves = (M_ / 32) * (N_ / 64);
      gemm_bf16_kernel<2, 4><<<(waves + 7) / 8, 256, 0, stream>>>(
          A, W, bias, resid, oF, oB, M_, N_, K_, op);
    } else {  // N_ == 32
      int waves = (M_ / 32) * (N_ / 32);
      gemm_bf16_kernel<2, 2><<<(waves + 7) / 8, 256, 0, stream>>>(
          A, W, bias, resid, oF, oB, M_, N_, K_, op);
    }
  };

  // Weight + value conversion to bf16
  cvt(qkv_w, wqkv, 768 * 256);
  cvt(proj_w, wproj, 256 * 256);
  cvt(off_w, woff, 64 * 256);
  cvt(aw_w, waw, 32 * 256);
  cvt(vproj_w, wvprj, 256 * 256);
  cvt(oproj_w, woprj, 256 * 256);
  cvt(fc1_w, wfc1, 1024 * 256);
  cvt(fc2_w, wfc2, 256 * 1024);
  cvt(value, val_b, M * C);

  // --- Stage 1: x1 = x + proj(attn(ln1(x))) ---
  ln_bf16_kernel<<<M / 8, 256, 0, stream>>>(x, ln1_w, ln1_b, ln_b, M);
  gemm(ln_b, wqkv, nullptr, nullptr, nullptr, qkv_b, M, 768, C, 0);
  vt_kernel<<<(M * C) / 256, 256, 0, stream>>>(qkv_b, vt_b);
  attn_kernel<<<4096 / 8, 256, 0, stream>>>(qkv_b, vt_b, ao_b);
  gemm(ao_b, wproj, proj_b, x, x1, nullptr, M, C, C, 0);

  // --- Stage 2: x2 = x1 + oproj(deform(ln2(x1))) ---
  ln_bf16_kernel<<<M / 8, 256, 0, stream>>>(x1, ln2_w, ln2_b, ln_b, M);
  gemm(val_b, wvprj, vproj_b, nullptr, vdef, nullptr, M, C, C, 0);
  gemm(ln_b, woff, off_b, nullptr, offf, nullptr, M, 64, C, 0);
  gemm(ln_b, waw, aw_b, nullptr, awlf, nullptr, M, 32, C, 0);
  deform_kernel<<<(M * 8) / 8, 256, 0, stream>>>(ref, vdef, offf, awlf, ao_b);
  gemm(ao_b, woprj, oproj_b, x1, x2, nullptr, M, C, C, 0);

  // --- Stage 3: out = x2 + fc2(gelu(fc1(ln3(x2)))) ---
  ln_bf16_kernel<<<M / 8, 256, 0, stream>>>(x2, ln3_w, ln3_b, ln_b, M);
  gemm(ln_b, wfc1, fc1_b, nullptr, nullptr, h_b, M, HID, C, 1);
  gemm(h_b, wfc2, fc2_b, x2, (float*)d_out, nullptr, M, C, HID, 0);
}
